// AttnBlock_76897094468467
// MI455X (gfx1250) — matrix operand order repaired and checked by reading, not compiled, not run on hardware
//
#include <hip/hip_runtime.h>

// ---------------------------------------------------------------------------
// AttnBlock (channel attention) for MI455X / gfx1250.
// wave32 + v_wmma_f32_16x16x32_f16 + Tensor Data Mover LDS staging.
// ---------------------------------------------------------------------------

#define C_DIM   512
#define N_PIX   4096
#define BATCH   16
#define GROUPS  32
#define CPG     16          // channels per group = C/GROUPS
#define EPS_GN  1e-6f
#define LDK     32          // K-step per WMMA stage (f16 K=32)
#define LDS_STR 40          // 32 + 8 pad halfs: 80B row, 16B aligned, conflict-free
#define EPI_STR 136         // epilogue f16 tile stride: 272B rows, 16B aligned
#define EPF_STR 132         // epilogue f32 tile stride: 528B rows, 16B aligned

typedef __attribute__((ext_vector_type(8)))  _Float16 v8h;
typedef __attribute__((ext_vector_type(16))) _Float16 v16h;
typedef __attribute__((ext_vector_type(8)))  float    v8f;
typedef __attribute__((ext_vector_type(4)))  unsigned int u32x4;
typedef __attribute__((ext_vector_type(8)))  int      i32x8;
typedef __attribute__((ext_vector_type(4)))  int      i32x4;

union Frag16 { v16h v; v8h h[2]; };
union HPair  { _Float16 h[2]; unsigned u; };

#if defined(__gfx1250__) && __has_builtin(__builtin_amdgcn_tensor_load_to_lds)
#define HAVE_TDM 1
#else
#define HAVE_TDM 0
#endif

// ---------------------------------------------------------------------------
// TDM: DMA a [rows x 32] f16 tile (row stride ld_elems halfs) from global
// into LDS, inserting 16B of pad every 64B so the LDS layout is exactly
// [row][LDS_STR] halfs.  pad_interval=3 -> 16 DWORDs (=32 halfs) between pads,
// pad_amount=3 -> 4 DWORDs (=8 halfs) of pad.  data_size=1 -> 2 bytes.
// Wave-level op (EXEC ignored), tracked on TENSORcnt.
// ---------------------------------------------------------------------------
__device__ __forceinline__
void tdm_load_tile(const _Float16* gsrc, unsigned lds_byte_off,
                   unsigned rows, unsigned ld_elems) {
#if HAVE_TDM
    unsigned long long ga = (unsigned long long)(uintptr_t)gsrc;
    u32x4 g0;
    g0[0] = 1u;                                         // count=1, user descriptor
    g0[1] = lds_byte_off;                               // lds_addr [63:32]
    g0[2] = (unsigned)(ga & 0xffffffffu);               // global_addr [95:64]
    g0[3] = (unsigned)((ga >> 32) & 0x01ffffffu)        // global_addr [120:96]
            | (2u << 30);                               // type=2 ("image")
    const unsigned td0 = 32u;                           // tensor_dim0 == tile_dim0
    i32x8 g1;
    g1[0] = (int)((1u << 16)                            // data_size = 2B
                  | (1u << 20)                          // pad_enable
                  | (3u << 22)                          // pad_interval: 16 DWORDs
                  | (3u << 25));                        // pad_amount:   4 DWORDs
    g1[1] = (int)((td0 & 0xffffu) << 16);               // tensor_dim0 lo
    g1[2] = (int)((td0 >> 16) | ((rows & 0xffffu) << 16));   // dim0 hi | dim1 lo
    g1[3] = (int)((rows >> 16) | (32u << 16));          // dim1 hi | tile_dim0=32
    g1[4] = (int)(rows & 0xffffu);                      // tile_dim1=rows, tile_dim2=0
    g1[5] = (int)ld_elems;                              // tensor_dim0_stride [31:0]
    g1[6] = 0;                                          // stride0 hi | stride1 lo
    g1[7] = 0;                                          // stride1 hi
    i32x4 gz = {0, 0, 0, 0};
#if __clang_major__ >= 23
    i32x8 gz8 = {0, 0, 0, 0, 0, 0, 0, 0};
    __builtin_amdgcn_tensor_load_to_lds(g0, g1, gz, gz, gz8, 0);
#else
    __builtin_amdgcn_tensor_load_to_lds(g0, g1, gz, gz, 0);
#endif
#else
    (void)gsrc; (void)lds_byte_off; (void)rows; (void)ld_elems;
#endif
}

__device__ __forceinline__ void tdm_wait_all() {
#if HAVE_TDM
    __builtin_amdgcn_s_wait_tensorcnt(0);
#endif
}

// ---------------------------------------------------------------------------
// Kernel 1: GroupNorm statistics -> per-(b,c) affine  hn = x*scale + shift
// ---------------------------------------------------------------------------
__global__ __launch_bounds__(256)
void gn_stats_kernel(const float* __restrict__ x,
                     const float* __restrict__ gamma,
                     const float* __restrict__ beta,
                     float* __restrict__ scale,   // [B][C]
                     float* __restrict__ shift) { // [B][C]
    const int b = blockIdx.x >> 5;
    const int g = blockIdx.x & 31;
    const int tid = threadIdx.x;
    const float* xg = x + ((size_t)b * C_DIM + (size_t)g * CPG) * N_PIX;
    const int total = CPG * N_PIX;    // 65536
    float s = 0.f, ss = 0.f;
    for (int i = tid; i < total; i += 256) {
        float v = xg[i];
        s += v; ss += v * v;
    }
    __shared__ float red0[256];
    __shared__ float red1[256];
    red0[tid] = s; red1[tid] = ss;
    __syncthreads();
    for (int off = 128; off > 0; off >>= 1) {
        if (tid < off) { red0[tid] += red0[tid + off]; red1[tid] += red1[tid + off]; }
        __syncthreads();
    }
    if (tid < CPG) {
        const float inv_n = 1.0f / (float)total;
        float mean = red0[0] * inv_n;
        float var  = red1[0] * inv_n - mean * mean;
        float rstd = rsqrtf(var + EPS_GN);
        int c = g * CPG + tid;
        float sc = rstd * gamma[c];
        scale[b * C_DIM + c] = sc;
        shift[b * C_DIM + c] = beta[c] - mean * sc;
    }
}

// ---------------------------------------------------------------------------
// Kernel 2: convert the four 512x512 f32 weight matrices to f16 (once per call)
// ---------------------------------------------------------------------------
__global__ __launch_bounds__(256)
void cvt_w_kernel(const float* __restrict__ Wq, const float* __restrict__ Wk,
                  const float* __restrict__ Wv, const float* __restrict__ Wo,
                  _Float16* __restrict__ w16) {  // [4][C][C]
    const int per = C_DIM * C_DIM;
    int idx = blockIdx.x * 256 + threadIdx.x;
    int m = idx / per, r = idx - m * per;
    const float* s = (m == 0) ? Wq : (m == 1) ? Wk : (m == 2) ? Wv : Wo;
    w16[idx] = (_Float16)s[r];
}

// ---------------------------------------------------------------------------
// Shared WMMA wave-level core: one K=32 step over a 128x128 block tile.
// 8 waves arranged 4(M) x 2(N); each wave owns 32x64 = 2x4 WMMA 16x16 tiles.
// sA: [128][LDS_STR] halfs, row-major M x K.   sB: [128][LDS_STR], N x K.
// ---------------------------------------------------------------------------
__device__ __forceinline__
void wave_mma_step(const _Float16* sA, const _Float16* sB,
                   int wm, int wn, int lane, v8f acc[2][4]) {
    Frag16 a[2], bf[4];
    const int r16 = lane & 15;
    const int hi  = lane >> 4;
#pragma unroll
    for (int tm = 0; tm < 2; ++tm) {
        // ISA 16-bit A 16x32: lane<16 -> K{0..7,16..23}, lane>=16 -> K{8..15,24..31}
        const _Float16* p = sA + (wm * 32 + tm * 16 + r16) * LDS_STR + hi * 8;
        a[tm].h[0] = *reinterpret_cast<const v8h*>(p);
        a[tm].h[1] = *reinterpret_cast<const v8h*>(p + 16);
    }
#pragma unroll
    for (int tn = 0; tn < 4; ++tn) {
        // ISA 16-bit B 32x16: N = lane%16, lane<16 -> K{0..7,16..23}, lane>=16 -> K{8..15,24..31}
        const _Float16* p = sB + (wn * 64 + tn * 16 + r16) * LDS_STR + hi * 8;
        bf[tn].h[0] = *reinterpret_cast<const v8h*>(p);
        bf[tn].h[1] = *reinterpret_cast<const v8h*>(p + 16);
    }
#pragma unroll
    for (int tm = 0; tm < 2; ++tm)
#pragma unroll
        for (int tn = 0; tn < 4; ++tn)
            acc[tm][tn] = __builtin_amdgcn_wmma_f32_16x16x32_f16(
                false, a[tm].v, false, bf[tn].v, (short)0, acc[tm][tn],
                false, false);
}

// VALU fallback: stage a 128x32 f16 tile from row-major global into LDS.
__device__ __forceinline__
void stage_rowmajor16(const _Float16* __restrict__ g, int ld,
                      _Float16* __restrict__ s, int tid) {
#pragma unroll
    for (int i = 0; i < 2; ++i) {
        int chunk = tid + i * 256;
        int row = chunk >> 2;
        int seg = (chunk & 3) * 8;
        *reinterpret_cast<v8h*>(&s[row * LDS_STR + seg]) =
            *reinterpret_cast<const v8h*>(&g[(size_t)row * ld + seg]);
    }
}

// A-tile loader: TDM (one 16-row slice per wave) or VALU fallback.
// g must already point at the tile origin (row m0, col k0).
__device__ __forceinline__
void load_a_tile(const _Float16* __restrict__ g, int ld,
                 _Float16* __restrict__ s, int tid, int wave) {
#if HAVE_TDM
    (void)tid;
    tdm_load_tile(g + (size_t)(wave * 16) * ld,
                  (unsigned)(uintptr_t)&s[wave * 16 * LDS_STR], 16u, (unsigned)ld);
#else
    (void)wave;
    stage_rowmajor16(g, ld, s, tid);
#endif
}

// Stage a K(32) x N(128) f16 tile from row-major [K][ld] global, transposed
// into LDS as [n][LDS_STR].  Two adjacent k rows per thread so each (k,k+1)
// half pair lands as one packed ds_store_b32.
__device__ __forceinline__
void stage_transpose16(const _Float16* __restrict__ g, int ld,
                       _Float16* __restrict__ s, int tid) {
    int kp = tid >> 4;             // 0..15: k pair
    int nl = (tid & 15) * 8;       // 0..120
    v8h r0 = *reinterpret_cast<const v8h*>(&g[(size_t)(2 * kp) * ld + nl]);
    v8h r1 = *reinterpret_cast<const v8h*>(&g[(size_t)(2 * kp + 1) * ld + nl]);
#pragma unroll
    for (int j = 0; j < 8; ++j) {
        HPair p;
        p.h[0] = r0[j]; p.h[1] = r1[j];
        *reinterpret_cast<unsigned*>(&s[(nl + j) * LDS_STR + 2 * kp]) = p.u;
    }
}

// Epilogue for f16 outputs: bounce the 128x128 tile through LDS, then flush
// with coalesced global b128 stores (8 per thread). bias may be nullptr.
__device__ __forceinline__
void epilogue_f16(v8f acc[2][4], const float* bias, int m0,
                  _Float16* sE, _Float16* __restrict__ ob, int ldo,
                  int wm, int wn, int lane, int tid) {
    const int r16 = lane & 15, hi = lane >> 4;
#pragma unroll
    for (int tm = 0; tm < 2; ++tm)
#pragma unroll
        for (int tn = 0; tn < 4; ++tn)
#pragma unroll
            for (int r = 0; r < 8; ++r) {
                int lr = wm * 32 + tm * 16 + r + hi * 8;   // 0..127
                int lc = wn * 64 + tn * 16 + r16;          // 0..127
                float v = acc[tm][tn][r] + (bias ? bias[m0 + lr] : 0.0f);
                sE[lr * EPI_STR + lc] = (_Float16)v;
            }
    __syncthreads();
#pragma unroll
    for (int i = 0; i < 8; ++i) {
        int chunk = tid + i * 256;        // 2048 chunks of 8 halfs
        int lr = chunk >> 4;
        int seg = (chunk & 15) * 8;
        *reinterpret_cast<v8h*>(&ob[(size_t)lr * ldo + seg]) =
            *reinterpret_cast<const v8h*>(&sE[lr * EPI_STR + seg]);
    }
}

// ---------------------------------------------------------------------------
// Kernel 3: fused GroupNorm-apply + q/k/v projection GEMM.
// out[mat][b][o][n] = sum_c W[mat][o][c] * (x[b][c][n]*scale+shift) + bias[o]
// grid: (N/128, M/128, B*3)
// ---------------------------------------------------------------------------
__global__ __launch_bounds__(256)
void qkv_gemm_kernel(const float* __restrict__ x,
                     const _Float16* __restrict__ w16,   // [3][C][C]
                     const float* __restrict__ bq,
                     const float* __restrict__ bk,
                     const float* __restrict__ bv,
                     const float* __restrict__ scale,
                     const float* __restrict__ shift,
                     _Float16* __restrict__ q16,
                     _Float16* __restrict__ k16,
                     _Float16* __restrict__ v16o) {
    __shared__ __attribute__((aligned(16))) _Float16 sA[128 * LDS_STR];
    __shared__ __attribute__((aligned(16))) _Float16 sB[128 * LDS_STR];
    __shared__ __attribute__((aligned(16))) _Float16 sE[128 * EPI_STR];
    const int tid  = threadIdx.x;
    const int lane = tid & 31;           // wave32
    const int wave = tid >> 5;
    const int wm = wave & 3, wn = wave >> 2;
    const int n0 = blockIdx.x * 128;
    const int m0 = blockIdx.y * 128;
    const int b   = blockIdx.z / 3;
    const int mat = blockIdx.z % 3;
    const _Float16* W = w16 + (size_t)mat * C_DIM * C_DIM;
    const float* bias = (mat == 0) ? bq : (mat == 1) ? bk : bv;
    _Float16* out = (mat == 0) ? q16 : (mat == 1) ? k16 : v16o;
    const float* xb  = x + (size_t)b * C_DIM * N_PIX;
    const float* scb = scale + b * C_DIM;
    const float* shb = shift + b * C_DIM;

    v8f acc[2][4] = {};

    for (int k0 = 0; k0 < C_DIM; k0 += LDK) {
        load_a_tile(W + (size_t)m0 * C_DIM + k0, C_DIM, sA, tid, wave);
        // Normalize x on the fly, f32 -> f16, transpose into [n][k].
        // Two adjacent c rows per chunk -> packed ds_store_b32.
#pragma unroll
        for (int i = 0; i < 2; ++i) {
            int chunk = tid + i * 256;   // 512 chunks of 2(k) x 4(n)
            int kp = chunk >> 5;         // 0..15
            int nl = (chunk & 31) * 4;   // 0..124
            int c0 = k0 + 2 * kp;
            float4 xa = *reinterpret_cast<const float4*>(
                &xb[(size_t)c0 * N_PIX + n0 + nl]);
            float4 xc = *reinterpret_cast<const float4*>(
                &xb[(size_t)(c0 + 1) * N_PIX + n0 + nl]);
            float s0 = scb[c0], h0 = shb[c0];
            float s1 = scb[c0 + 1], h1 = shb[c0 + 1];
            HPair p0, p1, p2, p3;
            p0.h[0] = (_Float16)(xa.x * s0 + h0); p0.h[1] = (_Float16)(xc.x * s1 + h1);
            p1.h[0] = (_Float16)(xa.y * s0 + h0); p1.h[1] = (_Float16)(xc.y * s1 + h1);
            p2.h[0] = (_Float16)(xa.z * s0 + h0); p2.h[1] = (_Float16)(xc.z * s1 + h1);
            p3.h[0] = (_Float16)(xa.w * s0 + h0); p3.h[1] = (_Float16)(xc.w * s1 + h1);
            unsigned* dst = reinterpret_cast<unsigned*>(&sB[nl * LDS_STR + 2 * kp]);
            dst[0]                    = p0.u;
            dst[(LDS_STR / 2)]        = p1.u;
            dst[(LDS_STR / 2) * 2]    = p2.u;
            dst[(LDS_STR / 2) * 3]    = p3.u;
        }
        if (k0 + LDK < C_DIM)  // global_prefetch_b8 of the next x tile
            __builtin_prefetch(&xb[(size_t)(k0 + LDK + (tid >> 3)) * N_PIX + n0], 0, 3);
        tdm_wait_all();
        __syncthreads();
        wave_mma_step(sA, sB, wm, wn, lane, acc);
        __syncthreads();
    }

    _Float16* ob = out + (size_t)b * C_DIM * N_PIX + (size_t)m0 * N_PIX + n0;
    epilogue_f16(acc, bias, m0, sE, ob, N_PIX, wm, wn, lane, tid);
}

// ---------------------------------------------------------------------------
// Kernel 4: scores[b][c][d] = C^-0.5 * sum_n q[b][c][n] * k[b][d][n]
// Both tiles go through the TDM with double-buffered LDS (in-order TENSORcnt:
// wait<=2 keeps the next pair in flight while the current pair is consumed).
// grid: (512/128, 512/128, B)
// ---------------------------------------------------------------------------
__global__ __launch_bounds__(256)
void scores_gemm_kernel(const _Float16* __restrict__ q16,
                        const _Float16* __restrict__ k16,
                        float* __restrict__ scores) {
    __shared__ __attribute__((aligned(16))) _Float16 sA[2][128 * LDS_STR];
    __shared__ __attribute__((aligned(16))) _Float16 sB[2][128 * LDS_STR];
    const int tid  = threadIdx.x;
    const int lane = tid & 31;
    const int wave = tid >> 5;
    const int wm = wave & 3, wn = wave >> 2;
    const int n0 = blockIdx.x * 128;   // d
    const int m0 = blockIdx.y * 128;   // c
    const int b  = blockIdx.z;
    const _Float16* qb = q16 + (size_t)b * C_DIM * N_PIX + (size_t)(m0 + wave * 16) * N_PIX;
    const _Float16* kb = k16 + (size_t)b * C_DIM * N_PIX + (size_t)(n0 + wave * 16) * N_PIX;

    v8f acc[2][4] = {};

#if HAVE_TDM
    const unsigned ldsA0 = (unsigned)(uintptr_t)&sA[0][wave * 16 * LDS_STR];
    const unsigned ldsA1 = (unsigned)(uintptr_t)&sA[1][wave * 16 * LDS_STR];
    const unsigned ldsB0 = (unsigned)(uintptr_t)&sB[0][wave * 16 * LDS_STR];
    const unsigned ldsB1 = (unsigned)(uintptr_t)&sB[1][wave * 16 * LDS_STR];
    tdm_load_tile(qb, ldsA0, 16u, N_PIX);
    tdm_load_tile(kb, ldsB0, 16u, N_PIX);
    int cur = 0;
    for (int k0 = 0; k0 < N_PIX; k0 += LDK) {
        if (k0 + LDK < N_PIX) {  // keep next pair in flight
            tdm_load_tile(qb + k0 + LDK, cur ? ldsA0 : ldsA1, 16u, N_PIX);
            tdm_load_tile(kb + k0 + LDK, cur ? ldsB0 : ldsB1, 16u, N_PIX);
            __builtin_amdgcn_s_wait_tensorcnt(2);   // current pair complete
        } else {
            __builtin_amdgcn_s_wait_tensorcnt(0);
        }
        __syncthreads();
        wave_mma_step(sA[cur], sB[cur], wm, wn, lane, acc);
        __syncthreads();
        cur ^= 1;
    }
#else
    for (int k0 = 0; k0 < N_PIX; k0 += LDK) {
        stage_rowmajor16(qb - (size_t)(wave * 16) * N_PIX + k0, N_PIX, sA[0], tid);
        stage_rowmajor16(kb - (size_t)(wave * 16) * N_PIX + k0, N_PIX, sB[0], tid);
        __syncthreads();
        wave_mma_step(sA[0], sB[0], wm, wn, lane, acc);
        __syncthreads();
    }
#endif

    const float sc = 0.044194173824159216f;  // 512^-0.5
    float* sb = scores + (size_t)b * C_DIM * C_DIM;
    const int r16 = lane & 15, hi = lane >> 4;
#pragma unroll
    for (int tm = 0; tm < 2; ++tm)
#pragma unroll
        for (int tn = 0; tn < 4; ++tn)
#pragma unroll
            for (int r = 0; r < 8; ++r) {
                int M = m0 + wm * 32 + tm * 16 + r + hi * 8;
                int N = n0 + wn * 64 + tn * 16 + r16;
                sb[(size_t)M * C_DIM + N] = acc[tm][tn][r] * sc;
            }
}

// ---------------------------------------------------------------------------
// Kernel 5: row softmax over d (512), output f16 attn. One block per row.
// ---------------------------------------------------------------------------
__global__ __launch_bounds__(256)
void softmax_kernel(const float* __restrict__ scores,
                    _Float16* __restrict__ attn16) {
    const int row = blockIdx.x;                    // b*512 + c
    const float* sr = scores + (size_t)row * C_DIM;
    const int tid = threadIdx.x;
    float v0 = sr[tid], v1 = sr[tid + 256];
    __shared__ float red[256];
    red[tid] = fmaxf(v0, v1);
    __syncthreads();
    for (int off = 128; off > 0; off >>= 1) {
        if (tid < off) red[tid] = fmaxf(red[tid], red[tid + off]);
        __syncthreads();
    }
    float mx = red[0];
    __syncthreads();
    float e0 = __expf(v0 - mx), e1 = __expf(v1 - mx);
    red[tid] = e0 + e1;
    __syncthreads();
    for (int off = 128; off > 0; off >>= 1) {
        if (tid < off) red[tid] += red[tid + off];
        __syncthreads();
    }
    float inv = 1.0f / red[0];
    _Float16* ar = attn16 + (size_t)row * C_DIM;
    ar[tid]       = (_Float16)(e0 * inv);
    ar[tid + 256] = (_Float16)(e1 * inv);
}

// ---------------------------------------------------------------------------
// Kernel 6: av[b][c][n] = sum_d attn[b][c][d] * v[b][d][n]
// grid: (N/128, 512/128, B)
// ---------------------------------------------------------------------------
__global__ __launch_bounds__(256)
void av_gemm_kernel(const _Float16* __restrict__ attn16,
                    const _Float16* __restrict__ v16,
                    _Float16* __restrict__ av16) {
    __shared__ __attribute__((aligned(16))) _Float16 sA[128 * LDS_STR];
    __shared__ __attribute__((aligned(16))) _Float16 sB[128 * LDS_STR];
    __shared__ __attribute__((aligned(16))) _Float16 sE[128 * EPI_STR];
    const int tid  = threadIdx.x;
    const int lane = tid & 31;
    const int wave = tid >> 5;
    const int wm = wave & 3, wn = wave >> 2;
    const int n0 = blockIdx.x * 128;
    const int m0 = blockIdx.y * 128;
    const int b  = blockIdx.z;
    const _Float16* ab = attn16 + (size_t)b * C_DIM * C_DIM;
    const _Float16* vb = v16 + (size_t)b * C_DIM * N_PIX;

    v8f acc[2][4] = {};

    for (int k0 = 0; k0 < C_DIM; k0 += LDK) {
        load_a_tile(ab + (size_t)m0 * C_DIM + k0, C_DIM, sA, tid, wave);
        stage_transpose16(vb + (size_t)k0 * N_PIX + n0, N_PIX, sB, tid);
        if (k0 + LDK < C_DIM)
            __builtin_prefetch(&vb[(size_t)(k0 + LDK + (tid >> 3)) * N_PIX + n0], 0, 3);
        tdm_wait_all();
        __syncthreads();
        wave_mma_step(sA, sB, wm, wn, lane, acc);
        __syncthreads();
    }

    _Float16* ob = av16 + (size_t)b * C_DIM * N_PIX + (size_t)m0 * N_PIX + n0;
    epilogue_f16(acc, nullptr, m0, sE, ob, N_PIX, wm, wn, lane, tid);
}

// ---------------------------------------------------------------------------
// Kernel 7: y[b][o][n] = x[b][o][n] + bo[o] + sum_c Wo[o][c] * av[b][c][n]
// f32 epilogue bounced through LDS in two 64x128 passes; bias + residual are
// folded into the vectorized (float4) read-back.
// grid: (N/128, 512/128, B)
// ---------------------------------------------------------------------------
__global__ __launch_bounds__(256)
void final_gemm_kernel(const _Float16* __restrict__ wo16,  // [C][C] f16
                       const _Float16* __restrict__ av16,
                       const float* __restrict__ bo,
                       const float* __restrict__ x,
                       float* __restrict__ y) {
    __shared__ __attribute__((aligned(16))) _Float16 sA[128 * LDS_STR];
    __shared__ __attribute__((aligned(16))) _Float16 sB[128 * LDS_STR];
    __shared__ __attribute__((aligned(16))) float    sF[64 * EPF_STR];
    const int tid  = threadIdx.x;
    const int lane = tid & 31;
    const int wave = tid >> 5;
    const int wm = wave & 3, wn = wave >> 2;
    const int n0 = blockIdx.x * 128;
    const int m0 = blockIdx.y * 128;
    const int b  = blockIdx.z;
    const _Float16* avb = av16 + (size_t)b * C_DIM * N_PIX;

    v8f acc[2][4] = {};

    for (int k0 = 0; k0 < C_DIM; k0 += LDK) {
        load_a_tile(wo16 + (size_t)m0 * C_DIM + k0, C_DIM, sA, tid, wave);
        stage_transpose16(avb + (size_t)k0 * N_PIX + n0, N_PIX, sB, tid);
        if (k0 + LDK < C_DIM)
            __builtin_prefetch(&avb[(size_t)(k0 + LDK + (tid >> 3)) * N_PIX + n0], 0, 3);
        tdm_wait_all();
        __syncthreads();
        wave_mma_step(sA, sB, wm, wn, lane, acc);
        __syncthreads();
    }

    const float* xb = x + (size_t)b * C_DIM * N_PIX;
    float* yb = y + (size_t)b * C_DIM * N_PIX;
    const int r16 = lane & 15, hi = lane >> 4;
#pragma unroll
    for (int p = 0; p < 2; ++p) {       // p == tm: 64 output rows per pass
#pragma unroll
        for (int tn = 0; tn < 4; ++tn)
#pragma unroll
            for (int r = 0; r < 8; ++r) {
                int lr = wm * 16 + r + hi * 8;           // 0..63
                int lc = wn * 64 + tn * 16 + r16;        // 0..127
                sF[lr * EPF_STR + lc] = acc[p][tn][r];
            }
        __syncthreads();
#pragma unroll
        for (int i = 0; i < 8; ++i) {
            int chunk = tid + i * 256;   // 2048 chunks of 4 floats
            int lr = chunk >> 5;         // 0..63
            int seg = (chunk & 31) * 4;  // 0..124
            int M = m0 + (lr >> 4) * 32 + p * 16 + (lr & 15);
            size_t gi = (size_t)M * N_PIX + n0 + seg;
            float4 xr = *reinterpret_cast<const float4*>(&xb[gi]);
            float bb = bo[M];
            const float* sp = &sF[lr * EPF_STR + seg];
            float4 o;
            o.x = sp[0] + bb + xr.x;
            o.y = sp[1] + bb + xr.y;
            o.z = sp[2] + bb + xr.z;
            o.w = sp[3] + bb + xr.w;
            *reinterpret_cast<float4*>(&yb[gi]) = o;
        }
        __syncthreads();
    }
}

// ---------------------------------------------------------------------------
// Host-side launch
// ---------------------------------------------------------------------------
extern "C" void kernel_launch(void* const* d_in, const int* in_sizes, int n_in,
                              void* d_out, int out_size, void* d_ws, size_t ws_size,
                              hipStream_t stream) {
    const float* x     = (const float*)d_in[0];
    const float* gamma = (const float*)d_in[1];
    const float* beta  = (const float*)d_in[2];
    const float* Wq    = (const float*)d_in[3];
    const float* bq    = (const float*)d_in[4];
    const float* Wk    = (const float*)d_in[5];
    const float* bk    = (const float*)d_in[6];
    const float* Wv    = (const float*)d_in[7];
    const float* bv    = (const float*)d_in[8];
    const float* Wo    = (const float*)d_in[9];
    const float* bo    = (const float*)d_in[10];
    float* y = (float*)d_out;

    char* ws = (char*)d_ws;
    const size_t sz_ss    = (size_t)BATCH * C_DIM * sizeof(float);
    const size_t sz_w16   = (size_t)4 * C_DIM * C_DIM * sizeof(_Float16);
    const size_t sz_act16 = (size_t)BATCH * C_DIM * N_PIX * sizeof(_Float16);
    const size_t sz_sc    = (size_t)BATCH * C_DIM * C_DIM * sizeof(float);
    const size_t sz_at16  = (size_t)BATCH * C_DIM * C_DIM * sizeof(_Float16);

    size_t off = 0;
    float*    scale  = (float*)(ws + off);            off += sz_ss;
    float*    shift  = (float*)(ws + off);            off += sz_ss;
    _Float16* w16    = (_Float16*)(ws + off);         off += sz_w16;
    _Float16* q16    = (_Float16*)(ws + off);         off += sz_act16;
    _Float16* k16    = (_Float16*)(ws + off);         off += sz_act16;
    _Float16* v16    = (_Float16*)(ws + off);         off += sz_act16;
    _Float16* av16   = (_Float16*)(ws + off);         off += sz_act16;
    float*    scores = (float*)(ws + off);            off += sz_sc;
    _Float16* attn16 = (_Float16*)(ws + off);         off += sz_at16;
    (void)ws_size; (void)in_sizes; (void)n_in; (void)out_size;

    gn_stats_kernel<<<BATCH * GROUPS, 256, 0, stream>>>(x, gamma, beta, scale, shift);
    cvt_w_kernel<<<(4 * C_DIM * C_DIM) / 256, 256, 0, stream>>>(Wq, Wk, Wv, Wo, w16);
    qkv_gemm_kernel<<<dim3(N_PIX / 128, C_DIM / 128, BATCH * 3), 256, 0, stream>>>(
        x, w16, bq, bk, bv, scale, shift, q16, k16, v16);
    scores_gemm_kernel<<<dim3(C_DIM / 128, C_DIM / 128, BATCH), 256, 0, stream>>>(
        q16, k16, scores);
    softmax_kernel<<<BATCH * C_DIM, 256, 0, stream>>>(scores, attn16);
    av_gemm_kernel<<<dim3(N_PIX / 128, C_DIM / 128, BATCH), 256, 0, stream>>>(
        attn16, v16, av16);
    final_gemm_kernel<<<dim3(N_PIX / 128, C_DIM / 128, BATCH), 256, 0, stream>>>(
        w16 + (size_t)3 * C_DIM * C_DIM, av16, bo, x, y);
}
